// Sinkhorn_low_level_23819888623714
// MI455X (gfx1250) — matrix operand order repaired and checked by reading, not compiled, not run on hardware
//
#include <hip/hip_runtime.h>
#include <hip/hip_bf16.h>
#include <math.h>

// Sinkhorn OT: B=16, N=M=2048, D=256, eps=0.1, 50 iters w/ freeze-on-converge.
// d_out = [cost (B*M) | pi (B*N*M) | C (B*N*M)] f32.

#define BB   16
#define NN   2048
#define MM   2048
#define DD   256
#define EPSR 0.1f
#define IEPS 10.0f
#define NSEG 16         // n-split for column passes (4 cols/thread -> keep blocks up)

typedef _Float16 v16h __attribute__((ext_vector_type(16)));
typedef _Float16 v8h  __attribute__((ext_vector_type(8)));
typedef float    v8f  __attribute__((ext_vector_type(8)));

#if __has_builtin(__builtin_amdgcn_global_load_async_to_lds_b128) && \
    __has_builtin(__builtin_amdgcn_s_wait_asynccnt)
#define HAVE_ASYNC_LDS 1
typedef int int4v __attribute__((vector_size(16)));
typedef __attribute__((address_space(1))) int4v* gp4;   // global int4*
typedef __attribute__((address_space(3))) int4v* lp4;   // LDS int4*
#endif

// ---------------------------------------------------------------------------
// Phase A: per-row normalize + split f32 -> (f16 hi, f16 lo)
// ---------------------------------------------------------------------------
__global__ __launch_bounds__(256)
void normalize_rows(const float* __restrict__ src,
                    _Float16* __restrict__ dh, _Float16* __restrict__ dl) {
    const int row = blockIdx.x;          // one 256-elem row per block
    const int t   = threadIdx.x;
    const size_t base = (size_t)row * DD + t;
    float val = src[base];
    __shared__ float red[256];
    red[t] = val * val;
    __syncthreads();
    for (int s = 128; s > 0; s >>= 1) {
        if (t < s) red[t] += red[t + s];
        __syncthreads();
    }
    const float rn = 1.0f / fmaxf(sqrtf(red[0]), 1e-12f);
    const float xn = val * rn;
    const _Float16 hi = (_Float16)xn;
    dh[base] = hi;
    dl[base] = (_Float16)(xn - (float)hi);
}

// ---------------------------------------------------------------------------
// Phase A2: init u=0, v=0, log_mu=log(mu+1e-8), done=0
// ---------------------------------------------------------------------------
__global__ __launch_bounds__(256)
void init_state(const float* __restrict__ mu, float* __restrict__ u,
                float* __restrict__ v, float* __restrict__ lmu,
                int* __restrict__ done) {
    const int i = blockIdx.x * 256 + threadIdx.x;   // 32768 == B*N == B*M
    u[i]   = 0.0f;
    v[i]   = 0.0f;
    lmu[i] = logf(mu[i] + 1e-8f);
    if (i == 0) *done = 0;
}

// ---------------------------------------------------------------------------
// Phase B: C[b,n,m] = 1 - dot(xn, yn) via f16 WMMA with hi/lo split (3 wmma)
// block tile 128x128, 8 waves, wave tile 32x64 (2x4 wmma tiles), K-chunk 32
// ---------------------------------------------------------------------------
__device__ __forceinline__
void stage_tile(const _Float16* __restrict__ g, _Float16 (*s)[40],
                int baseRow, int kc, int tid) {
    const int r  = tid >> 1;             // 128 rows, 2 threads per row
    const int cb = (tid & 1) << 4;       // half-offset 0 or 16
    const _Float16* src = g + (size_t)(baseRow + r) * DD + kc + cb;
#ifdef HAVE_ASYNC_LDS
    __builtin_amdgcn_global_load_async_to_lds_b128(
        (gp4)(src),     (lp4)&s[r][cb],     0, 0);
    __builtin_amdgcn_global_load_async_to_lds_b128(
        (gp4)(src + 8), (lp4)&s[r][cb + 8], 0, 0);
#else
    *(v8h*)&s[r][cb]     = *(const v8h*)(src);
    *(v8h*)&s[r][cb + 8] = *(const v8h*)(src + 8);
#endif
}

__device__ __forceinline__
v16h load_fragA(_Float16 (*s)[40], int rbase, int lane) {
    // 16-bit A 16x32 layout: lane L -> row L&15; kb=(L>>4)*8; K = kb+h (h<8), kb+16+h-8
    const int r  = rbase + (lane & 15);
    const int kb = (lane >> 4) * 8;
    v8h a = *(const v8h*)&s[r][kb];
    v8h b = *(const v8h*)&s[r][kb + 16];
    return __builtin_shufflevector(a, b, 0,1,2,3,4,5,6,7,8,9,10,11,12,13,14,15);
}

__device__ __forceinline__
v16h load_fragB(_Float16 (*s)[40], int cbase, int lane) {
    // 16-bit B 32x16 layout: lane L -> col L&15; kb=(L>>4)*8; K = kb+h (h<8), kb+16+h-8
    const int c  = cbase + (lane & 15);
    const int kb = (lane >> 4) * 8;
    v8h a = *(const v8h*)&s[c][kb];
    v8h b = *(const v8h*)&s[c][kb + 16];
    return __builtin_shufflevector(a, b, 0,1,2,3,4,5,6,7,8,9,10,11,12,13,14,15);
}

__global__ __launch_bounds__(256)
void gemm_cost(const _Float16* __restrict__ xh, const _Float16* __restrict__ xl,
               const _Float16* __restrict__ yh, const _Float16* __restrict__ yl,
               float* __restrict__ Cout) {
    const int b  = blockIdx.z;
    const int n0 = blockIdx.y * 128;
    const int m0 = blockIdx.x * 128;
    const int tid  = threadIdx.x;
    const int lane = tid & 31;
    const int wid  = tid >> 5;
    const int wr   = wid & 3;    // 4 row groups of 32
    const int wc   = wid >> 2;   // 2 col groups of 64

    __shared__ _Float16 sXh[128][40];
    __shared__ _Float16 sXl[128][40];
    __shared__ _Float16 sYh[128][40];
    __shared__ _Float16 sYl[128][40];

    const int xBase = b * NN + n0;
    const int yBase = b * MM + m0;

    v8f acc[2][4];
    const v8f vz = {0.f,0.f,0.f,0.f,0.f,0.f,0.f,0.f};
#pragma unroll
    for (int i = 0; i < 2; ++i)
#pragma unroll
        for (int j = 0; j < 4; ++j) acc[i][j] = vz;

    for (int kc = 0; kc < DD; kc += 32) {
        stage_tile(xh, sXh, xBase, kc, tid);
        stage_tile(xl, sXl, xBase, kc, tid);
        stage_tile(yh, sYh, yBase, kc, tid);
        stage_tile(yl, sYl, yBase, kc, tid);
#ifdef HAVE_ASYNC_LDS
        __builtin_amdgcn_s_wait_asynccnt(0);
#endif
        __syncthreads();

        v16h aH[2], aL[2];
#pragma unroll
        for (int rt = 0; rt < 2; ++rt) {
            aH[rt] = load_fragA(sXh, wr * 32 + rt * 16, lane);
            aL[rt] = load_fragA(sXl, wr * 32 + rt * 16, lane);
        }
#pragma unroll
        for (int ct = 0; ct < 4; ++ct) {
            v16h bH = load_fragB(sYh, wc * 64 + ct * 16, lane);
            v16h bL = load_fragB(sYl, wc * 64 + ct * 16, lane);
#pragma unroll
            for (int rt = 0; rt < 2; ++rt) {
                acc[rt][ct] = __builtin_amdgcn_wmma_f32_16x16x32_f16(
                    false, aH[rt], false, bH, (short)0, acc[rt][ct], false, false);
                acc[rt][ct] = __builtin_amdgcn_wmma_f32_16x16x32_f16(
                    false, aH[rt], false, bL, (short)0, acc[rt][ct], false, false);
                acc[rt][ct] = __builtin_amdgcn_wmma_f32_16x16x32_f16(
                    false, aL[rt], false, bH, (short)0, acc[rt][ct], false, false);
            }
        }
        __syncthreads();
    }

    // C/D layout: lane L -> col L&15; VGPR r -> row r + (L>>4)*8
    const size_t cB = (size_t)b * NN * MM;
#pragma unroll
    for (int rt = 0; rt < 2; ++rt) {
#pragma unroll
        for (int ct = 0; ct < 4; ++ct) {
            const int nrow0 = n0 + wr * 32 + rt * 16 + (lane >> 4) * 8;
            const int mcol  = m0 + wc * 64 + ct * 16 + (lane & 15);
#pragma unroll
            for (int r = 0; r < 8; ++r) {
                Cout[cB + (size_t)(nrow0 + r) * MM + mcol] = 1.0f - acc[rt][ct][r];
            }
        }
    }
}

// ---------------------------------------------------------------------------
// Phase C: Sinkhorn iteration kernels (freeze once converged)
// ---------------------------------------------------------------------------
__global__ __launch_bounds__(256)
void sink_row(const float* __restrict__ C, float* __restrict__ u,
              const float* __restrict__ v, const float* __restrict__ lmu,
              float* __restrict__ errbuf, const int* __restrict__ done) {
    if (*done) return;
    const int n = blockIdx.x, b = blockIdx.y, t = threadIdx.x;
    const int ridx = b * NN + n;
    const size_t base = (size_t)ridx * MM;
    const float uo = u[ridx];
    const int m0 = t * 8;                           // 8 contiguous m per thread
    const float4 cA = *(const float4*)&C[base + m0];
    const float4 cB = *(const float4*)&C[base + m0 + 4];
    const float4 vA = *(const float4*)&v[b * MM + m0];
    const float4 vB = *(const float4*)&v[b * MM + m0 + 4];
    float tv[8];
    tv[0] = (uo + vA.x - cA.x) * IEPS;  tv[1] = (uo + vA.y - cA.y) * IEPS;
    tv[2] = (uo + vA.z - cA.z) * IEPS;  tv[3] = (uo + vA.w - cA.w) * IEPS;
    tv[4] = (uo + vB.x - cB.x) * IEPS;  tv[5] = (uo + vB.y - cB.y) * IEPS;
    tv[6] = (uo + vB.z - cB.z) * IEPS;  tv[7] = (uo + vB.w - cB.w) * IEPS;
    float mx = tv[0];
#pragma unroll
    for (int i = 1; i < 8; ++i) mx = fmaxf(mx, tv[i]);
    __shared__ float red[256];
    red[t] = mx; __syncthreads();
    for (int s = 128; s > 0; s >>= 1) {
        if (t < s) red[t] = fmaxf(red[t], red[t + s]);
        __syncthreads();
    }
    mx = red[0]; __syncthreads();
    float sum = 0.f;
#pragma unroll
    for (int i = 0; i < 8; ++i) sum += expf(tv[i] - mx);
    red[t] = sum; __syncthreads();
    for (int s = 128; s > 0; s >>= 1) {
        if (t < s) red[t] += red[t + s];
        __syncthreads();
    }
    if (t == 0) {
        const float lse = mx + logf(red[0]);
        const float un  = EPSR * (lmu[ridx] - lse) + uo;
        u[ridx] = un;
        errbuf[ridx] = fabsf(un - uo);
    }
}

__device__ __forceinline__
void lse_step(float val, float& mx, float& s) {
    if (val > mx) { s = s * expf(mx - val) + 1.0f; mx = val; }
    else          { s += expf(val - mx); }
}

__global__ __launch_bounds__(256)
void sink_col_partial(const float* __restrict__ C, const float* __restrict__ u,
                      const float* __restrict__ v, float* __restrict__ pmax,
                      float* __restrict__ psum, const int* __restrict__ done) {
    if (*done) return;
    const int b = blockIdx.z, seg = blockIdx.y;
    const int m0 = blockIdx.x * 1024 + threadIdx.x * 4;   // 4 cols per thread
    const float4 vo = *(const float4*)&v[b * MM + m0];
    const size_t cb = (size_t)b * NN * MM;
    float mx0 = -INFINITY, mx1 = -INFINITY, mx2 = -INFINITY, mx3 = -INFINITY;
    float s0 = 0.f, s1 = 0.f, s2 = 0.f, s3 = 0.f;
    for (int i = 0; i < NN / NSEG; ++i) {
        const int n = seg * (NN / NSEG) + i;
        const size_t off = cb + (size_t)n * MM + m0;
        __builtin_prefetch(&C[off + 8 * MM], 0, 1);
        const float un = u[b * NN + n];
        const float4 c = *(const float4*)&C[off];
        lse_step((un + vo.x - c.x) * IEPS, mx0, s0);
        lse_step((un + vo.y - c.y) * IEPS, mx1, s1);
        lse_step((un + vo.z - c.z) * IEPS, mx2, s2);
        lse_step((un + vo.w - c.w) * IEPS, mx3, s3);
    }
    const int o = (seg * BB + b) * MM + m0;
    *(float4*)&pmax[o] = make_float4(mx0, mx1, mx2, mx3);
    *(float4*)&psum[o] = make_float4(s0, s1, s2, s3);
}

__global__ __launch_bounds__(256)
void sink_col_combine(const float* __restrict__ pmax, const float* __restrict__ psum,
                      float* __restrict__ v, const int* __restrict__ done) {
    if (*done) return;
    const int b = blockIdx.y;
    const int m = blockIdx.x * 256 + threadIdx.x;
    float gmx = -INFINITY;
#pragma unroll
    for (int seg = 0; seg < NSEG; ++seg)
        gmx = fmaxf(gmx, pmax[(seg * BB + b) * MM + m]);
    float gs = 0.f;
#pragma unroll
    for (int seg = 0; seg < NSEG; ++seg) {
        const int o = (seg * BB + b) * MM + m;
        gs += psum[o] * expf(pmax[o] - gmx);
    }
    const float lse = gmx + logf(gs);
    const float log_nu = logf(1.0f / (float)MM + 1e-8f);
    const int idx = b * MM + m;
    v[idx] = EPSR * (log_nu - lse) + v[idx];
}

__global__ __launch_bounds__(1024)
void err_reduce(const float* __restrict__ errbuf, int* __restrict__ done) {
    if (*done) return;   // freeze semantics: checked AFTER both updates ran
    __shared__ float red[1024];
    const int t = threadIdx.x;
    float s = 0.f;
    for (int i = 0; i < 32; ++i) s += errbuf[t * 32 + i];   // fixed order
    red[t] = s; __syncthreads();
    for (int sz = 512; sz > 0; sz >>= 1) {
        if (t < sz) red[t] += red[t + sz];
        __syncthreads();
    }
    if (t == 0 && (red[0] / (float)BB) < 1e-5f) *done = 1;
}

// ---------------------------------------------------------------------------
// Phase D: pi = exp(M), cost[b,m] = sum_n pi*C (deterministic 2-level sum)
// ---------------------------------------------------------------------------
__global__ __launch_bounds__(256)
void final_partial(const float* __restrict__ C, const float* __restrict__ u,
                   const float* __restrict__ v, float* __restrict__ pi,
                   float* __restrict__ pcost) {
    const int b = blockIdx.z, seg = blockIdx.y;
    const int m0 = blockIdx.x * 1024 + threadIdx.x * 4;   // 4 cols per thread
    const float4 vo = *(const float4*)&v[b * MM + m0];
    const size_t cb = (size_t)b * NN * MM;
    float a0 = 0.f, a1 = 0.f, a2 = 0.f, a3 = 0.f;
    for (int i = 0; i < NN / NSEG; ++i) {
        const int n = seg * (NN / NSEG) + i;
        const size_t off = cb + (size_t)n * MM + m0;
        __builtin_prefetch(&C[off + 8 * MM], 0, 1);
        const float un = u[b * NN + n];
        const float4 c = *(const float4*)&C[off];
        float4 p;
        p.x = expf((un + vo.x - c.x) * IEPS);
        p.y = expf((un + vo.y - c.y) * IEPS);
        p.z = expf((un + vo.z - c.z) * IEPS);
        p.w = expf((un + vo.w - c.w) * IEPS);
        *(float4*)&pi[off] = p;
        a0 += p.x * c.x;  a1 += p.y * c.y;  a2 += p.z * c.z;  a3 += p.w * c.w;
    }
    *(float4*)&pcost[(seg * BB + b) * MM + m0] = make_float4(a0, a1, a2, a3);
}

__global__ __launch_bounds__(256)
void final_combine(const float* __restrict__ pcost, float* __restrict__ cost) {
    const int b = blockIdx.y;
    const int m = blockIdx.x * 256 + threadIdx.x;
    float s = 0.f;
#pragma unroll
    for (int seg = 0; seg < NSEG; ++seg) s += pcost[(seg * BB + b) * MM + m];
    cost[b * MM + m] = s;
}

// ---------------------------------------------------------------------------
extern "C" void kernel_launch(void* const* d_in, const int* in_sizes, int n_in,
                              void* d_out, int out_size, void* d_ws, size_t ws_size,
                              hipStream_t stream) {
    const float* x  = (const float*)d_in[0];   // [B,N,D]
    const float* y  = (const float*)d_in[1];   // [B,M,D]
    const float* mu = (const float*)d_in[2];   // [B,N]

    float* out  = (float*)d_out;
    float* cost = out;                                     // B*M
    float* pi   = out + (size_t)BB * MM;                   // B*N*M
    float* Cc   = pi  + (size_t)BB * NN * MM;              // B*N*M

    // workspace carve-up (~72 MB)
    _Float16* xh = (_Float16*)d_ws;
    _Float16* xl = xh + (size_t)BB * NN * DD;
    _Float16* yh = xl + (size_t)BB * NN * DD;
    _Float16* yl = yh + (size_t)BB * MM * DD;
    float* u    = (float*)(yl + (size_t)BB * MM * DD);
    float* v    = u    + BB * NN;
    float* lmu  = v    + BB * MM;
    float* errb = lmu  + BB * NN;
    float* pmax = errb + BB * NN;
    float* psum = pmax + NSEG * BB * MM;
    int*   done = (int*)(psum + NSEG * BB * MM);

    normalize_rows<<<BB * NN, 256, 0, stream>>>(x, xh, xl);
    normalize_rows<<<BB * MM, 256, 0, stream>>>(y, yh, yl);
    init_state<<<(BB * NN) / 256, 256, 0, stream>>>(mu, u, v, lmu, done);

    gemm_cost<<<dim3(MM / 128, NN / 128, BB), 256, 0, stream>>>(xh, xl, yh, yl, Cc);

    for (int it = 0; it < 50; ++it) {
        sink_row<<<dim3(NN, BB), 256, 0, stream>>>(Cc, u, v, lmu, errb, done);
        sink_col_partial<<<dim3(MM / 1024, NSEG, BB), 256, 0, stream>>>(Cc, u, v, pmax, psum, done);
        sink_col_combine<<<dim3(MM / 256, BB), 256, 0, stream>>>(pmax, psum, v, done);
        err_reduce<<<1, 1024, 0, stream>>>(errb, done);   // sets freeze for NEXT iter
    }

    final_partial<<<dim3(MM / 1024, NSEG, BB), 256, 0, stream>>>(Cc, u, v, pi, pmax /*pcost*/);
    final_combine<<<dim3(MM / 256, BB), 256, 0, stream>>>(pmax, cost);
}
